// KroneckerLinear_69028714381607
// MI455X (gfx1250) — matrix operand order repaired and checked by reading, not compiled, not run on hardware
//
#include <hip/hip_runtime.h>

// ---------------------------------------------------------------------------
// KroneckerLinear on MI455X (gfx1250, wave32, WMMA):
//   W[o,i] = sum_b a[b, o/512, i/512] * s[b, o%512, i%512]   (4096x4096)
//   out    = x @ W^T,  x:(8192,4096) f32, out:(8192,4096) f32
// Stage 1: materialize W in f16 into workspace (cheap: 0.27 GFLOP).
// Stage 2: f16 WMMA GEMM, f32 accumulate (275 GFLOP, compute-bound).
//          B tile copied global->LDS with GLOBAL_LOAD_ASYNC_TO_LDS_B128
//          (ASYNCcnt path) when the builtin is available.
// ---------------------------------------------------------------------------

typedef __attribute__((ext_vector_type(16))) _Float16 v16h;
typedef __attribute__((ext_vector_type(8)))  _Float16 v8h;
typedef __attribute__((ext_vector_type(4)))  _Float16 v4h;
typedef __attribute__((ext_vector_type(8)))  float    v8f;
typedef __attribute__((ext_vector_type(4)))  int      v4i;

#define IN_DIM  4096
#define OUT_DIM 4096
#define FACTOR  8
#define SDIM    512      // IN_DIM / FACTOR

#define BM 128
#define BN 128
#define BK 32
#define LDA 40           // 32 + 8 halves pad -> 80B row stride, conflict-free b128 reads
#define LDB 40

#if __has_builtin(__builtin_amdgcn_global_load_async_to_lds_b128) && \
    __has_builtin(__builtin_amdgcn_s_wait_asynccnt)
#define HAVE_ASYNC_LDS 1
#else
#define HAVE_ASYNC_LDS 0
#endif

#if HAVE_ASYNC_LDS
// Builtin signature (per clang diagnostic): (v4i AS1*, v4i AS3*, imm, imm).
// Generic->AS1: same 64-bit VA. Generic->AS3: low 32 bits are the LDS offset
// (per ISA flat addressing).
typedef __attribute__((address_space(1))) v4i as1_v4i;
typedef __attribute__((address_space(3))) v4i as3_v4i;
#define TO_GLOBAL(p) ((as1_v4i*)(uintptr_t)(p))
#define TO_LDS(p)    ((as3_v4i*)(unsigned int)(uintptr_t)(p))
#endif

// ---------------------------------------------------------------------------
// Stage 1: W[o][i] = sum_b a[b][o>>9][i>>9] * s[b][o&511][i&511], f16 row-major.
// One thread produces 4 consecutive i (same Kronecker cell since 512 % 4 == 0).
// ---------------------------------------------------------------------------
__global__ __launch_bounds__(256)
void build_weight(const float* __restrict__ a, const float* __restrict__ s,
                  _Float16* __restrict__ w) {
    const int idx = blockIdx.x * blockDim.x + threadIdx.x;   // 4 outputs each
    const int o  = idx >> 10;            // IN_DIM/4 = 1024 quads per row
    const int i0 = (idx & 1023) << 2;
    const int ar = o  >> 9, kr = o  & 511;
    const int ac = i0 >> 9, pc = i0 & 511;

    float acc0 = 0.f, acc1 = 0.f, acc2 = 0.f, acc3 = 0.f;
#pragma unroll
    for (int b = 0; b < FACTOR; ++b) {
        const float av = a[(b * FACTOR + ar) * FACTOR + ac];
        const float4 sv = *(const float4*)&s[((size_t)(b * SDIM + kr)) * SDIM + pc];
        acc0 += av * sv.x; acc1 += av * sv.y;
        acc2 += av * sv.z; acc3 += av * sv.w;
    }
    v4h hv = { (_Float16)acc0, (_Float16)acc1, (_Float16)acc2, (_Float16)acc3 };
    *(v4h*)&w[(size_t)o * IN_DIM + i0] = hv;
}

// ---------------------------------------------------------------------------
// Stage 2: out = X(f32) @ W(f16)^T via v_wmma_f32_16x16x32_f16.
// 256 threads = 8 wave32 per block; block tile 128x128, K-step 32.
// Wave grid 4(m) x 2(n): each wave owns 32x64 = 2x4 WMMA accumulators.
// ---------------------------------------------------------------------------
__global__ __launch_bounds__(256)
void kron_gemm(const float* __restrict__ X, const _Float16* __restrict__ W,
               float* __restrict__ out) {
    __shared__ _Float16 As[BM * LDA];   // [m][k], f16, padded stride
    __shared__ _Float16 Bs[BN * LDB];   // [n][k], f16, padded stride

    const int tid  = threadIdx.x;
    const int n0   = blockIdx.x * BN;
    const int m0   = blockIdx.y * BM;
    const int wid  = tid >> 5;
    const int lane = tid & 31;
    const int h    = lane >> 4;         // lane half (K-group select)
    const int ln   = lane & 15;         // row/col within fragment
    const int wm   = (wid & 3) * 32;    // wave M offset in block tile
    const int wn   = (wid >> 2) * 64;   // wave N offset in block tile

    v8f acc[2][4] = {};                 // 8x f32 16x16 accumulators

    for (int k0 = 0; k0 < IN_DIM; k0 += BK) {
        // ---- A tile: 128x32 f32 -> f16 into LDS (4 float4 per thread) ----
#pragma unroll
        for (int it = 0; it < 4; ++it) {
            const int slot = tid + it * 256;        // 0..1023 quads
            const int m    = slot >> 3;
            const int kq   = (slot & 7) << 2;
            const float4 xv = *(const float4*)&X[(size_t)(m0 + m) * IN_DIM + k0 + kq];
            v4h hv = { (_Float16)xv.x, (_Float16)xv.y, (_Float16)xv.z, (_Float16)xv.w };
            *(v4h*)&As[m * LDA + kq] = hv;
        }
        // ---- B tile: 128x32 f16 rows of W into LDS (2 x 16B per thread) ----
#pragma unroll
        for (int it = 0; it < 2; ++it) {
            const int slot = tid + it * 256;        // 0..511 octets
            const int n    = slot >> 2;
            const int ko   = (slot & 3) << 3;
            const _Float16* gsrc = &W[(size_t)(n0 + n) * IN_DIM + k0 + ko];
            _Float16*       ldst = &Bs[n * LDB + ko];
#if HAVE_ASYNC_LDS
            // CDNA5 async DMA: global -> LDS without touching VGPR data path.
            __builtin_amdgcn_global_load_async_to_lds_b128(
                TO_GLOBAL(gsrc), TO_LDS(ldst), 0, 0);
#else
            *(v8h*)ldst = *(const v8h*)gsrc;
#endif
        }
        // Prefetch next K tile into near caches while this one is consumed.
        if (k0 + BK < IN_DIM) {
            __builtin_prefetch(&X[(size_t)(m0 + (tid >> 1)) * IN_DIM + k0 + BK], 0, 3);
            __builtin_prefetch(&W[(size_t)(n0 + (tid >> 1)) * IN_DIM + k0 + BK], 0, 3);
        }
#if HAVE_ASYNC_LDS
        __builtin_amdgcn_s_wait_asynccnt(0);   // B tile landed in LDS
#endif
        __syncthreads();

        // ---- Fragment loads per documented CDNA5 VGPR layouts ----
        v16h af[2], bf[4];
#pragma unroll
        for (int mf = 0; mf < 2; ++mf) {
            // A 16x32: lane = row ln, elems 0..7 at k=h*8, elems 8..15 at k=16+h*8
            const _Float16* p = &As[(wm + mf * 16 + ln) * LDA + h * 8];
            const v8h lo = *(const v8h*)p;
            const v8h hi = *(const v8h*)(p + 16);
#pragma unroll
            for (int i = 0; i < 8; ++i) { af[mf][i] = lo[i]; af[mf][8 + i] = hi[i]; }
        }
#pragma unroll
        for (int nf = 0; nf < 4; ++nf) {
            // B 32x16: lane = col ln, elems 0..7 at k=h*8, elems 8..15 at k=16+h*8
            const _Float16* p = &Bs[(wn + nf * 16 + ln) * LDB + h * 8];
            const v8h lo = *(const v8h*)p;
            const v8h hi = *(const v8h*)(p + 16);
#pragma unroll
            for (int i = 0; i < 8; ++i) { bf[nf][i] = lo[i]; bf[nf][8 + i] = hi[i]; }
        }

        // ---- 8 WMMA per K step ----
#pragma unroll
        for (int mf = 0; mf < 2; ++mf)
#pragma unroll
            for (int nf = 0; nf < 4; ++nf)
                acc[mf][nf] = __builtin_amdgcn_wmma_f32_16x16x32_f16(
                    false, af[mf], false, bf[nf],
                    (short)0, acc[mf][nf], false, false);

        __syncthreads();
    }

    // ---- Epilogue: C/D layout — VGPR r holds M = r + 8*(lane/16), N = lane%16 ----
#pragma unroll
    for (int mf = 0; mf < 2; ++mf)
#pragma unroll
        for (int nf = 0; nf < 4; ++nf) {
            const int n = n0 + wn + nf * 16 + ln;
#pragma unroll
            for (int r = 0; r < 8; ++r) {
                const int m = m0 + wm + mf * 16 + r + 8 * h;
                out[(size_t)m * OUT_DIM + n] = acc[mf][nf][r];
            }
        }
}

// ---------------------------------------------------------------------------
extern "C" void kernel_launch(void* const* d_in, const int* in_sizes, int n_in,
                              void* d_out, int out_size, void* d_ws, size_t ws_size,
                              hipStream_t stream) {
    const float* x = (const float*)d_in[0];   // (4,2048,4096) f32
    const float* a = (const float*)d_in[1];   // (8,8,8) f32
    const float* s = (const float*)d_in[2];   // (8,512,512) f32
    float*     out = (float*)d_out;           // (4,2048,4096) f32
    _Float16*    w = (_Float16*)d_ws;         // 4096*4096 f16 = 33.5 MB scratch

    // Stage 1: materialize Kronecker-sum weight in f16.
    build_weight<<<(OUT_DIM * IN_DIM) / (256 * 4), 256, 0, stream>>>(a, s, w);

    // Stage 2: tiled WMMA GEMM, M=8192, N=4096, K=4096.
    dim3 grid(OUT_DIM / BN, (4 * 2048) / BM);
    kron_gemm<<<grid, 256, 0, stream>>>(x, w, out);
}
